// RNN_14714557956636
// MI455X (gfx1250) — matrix operand order repaired and checked by reading, not compiled, not run on hardware
//
#include <hip/hip_runtime.h>
#include <hip/hip_bf16.h>
#include <stdint.h>

#define T_STEPS 512
#define BATCH   128
#define N_INP   256
#define N_HID   1024
#define N_OUT   128

typedef __attribute__((ext_vector_type(16))) __bf16 v16bf;
typedef __attribute__((ext_vector_type(8)))  float  v8f;

struct U4x2 { uint4 lo; uint4 hi; };
union ABCast { U4x2 u; v16bf v; };
union PackCast { uint16_t s[16]; v16bf v; };

__device__ __forceinline__ uint16_t f2bf(float f) {
  union { float f; uint32_t u; } c; c.f = f;
  uint32_t r = c.u + 0x7FFFu + ((c.u >> 16) & 1u);   // round-to-nearest-even
  return (uint16_t)(r >> 16);
}

// A operand (16x32 bf16): lane<16 holds row M=lane, K {0..15} contiguous;
// lane>=16 same rows, K {16..31}.
__device__ __forceinline__ v16bf load_A(const uint16_t* base /*16B aligned*/) {
  ABCast a;
  const uint4* p = (const uint4*)base;
  a.u.lo = p[0];        // K  +0..7   (8 bf16)
  a.u.hi = p[1];        // K  +8..15  (8 bf16)
  return a.v;
}
// B operand (32x16 bf16): VGPR v holds rows k={2v,2v+1} striped across lanes;
// per lane (col N=lane&15) K is CONTIGUOUS: K = kgrp*16 .. kgrp*16+15.
__device__ __forceinline__ v16bf load_B(const uint16_t* base /*16B aligned*/) {
  ABCast b;
  const uint4* p = (const uint4*)base;
  b.u.lo = p[0];        // K +0..7
  b.u.hi = p[1];        // K +8..15
  return b.v;
}

// ---------------- prep: f32 -> bf16 weights, zero state staging + barrier ----
__global__ void prep_kernel(const float* __restrict__ W_ih,
                            const float* __restrict__ W_hh,
                            uint16_t* whh_bf, uint16_t* wih_bf,
                            uint16_t* s0, unsigned* bar) {
  int i = blockIdx.x * blockDim.x + threadIdx.x;
  int stride = gridDim.x * blockDim.x;
  for (int k = i; k < N_HID * N_HID; k += stride) whh_bf[k] = f2bf(W_hh[k]);
  for (int k = i; k < N_HID * N_INP; k += stride) wih_bf[k] = f2bf(W_ih[k]);
  for (int k = i; k < BATCH * N_HID; k += stride) s0[k] = 0;   // h0 = 0
  if (i < 2) bar[i] = 0;
}

// ---------------- persistent recurrence kernel ------------------------------
// 64 WGs = 8 batch-tiles x 8 hidden-tiles; 8 waves/WG; wave = one 16x16 D tile.
__launch_bounds__(256, 1)
__global__ void rnn_persist(const float* __restrict__ x,
                            const uint16_t* __restrict__ whh_bf,
                            const uint16_t* __restrict__ wih_bf,
                            const float* __restrict__ b_ih,
                            const float* __restrict__ b_hh,
                            uint16_t* s0, uint16_t* s1,
                            float* fin, unsigned* bar) {
  __shared__ uint16_t lds_state[16 * N_HID];  // 32 KB prev-state rows (batch tile)
  __shared__ uint16_t lds_x[16 * N_INP];      //  8 KB x[t] tile as bf16

  const int tid  = threadIdx.x;
  const int wave = tid >> 5;
  const int lane = tid & 31;
  const int bt   = blockIdx.x >> 3;   // batch tile 0..7
  const int ht   = blockIdx.x & 7;    // hidden tile 0..7
  const int mrow = lane & 15;         // A: M row; B/C/D: N column
  const int kgrp = lane >> 4;
  const int hcol = ht * 128 + wave * 16 + mrow;   // output hidden column
  const float bias = b_ih[hcol] + b_hh[hcol];
  const int nblocks = gridDim.x;

  const unsigned lds_state_base = (unsigned)(uintptr_t)(&lds_state[0]);
  uint16_t* sbuf0 = s0;
  uint16_t* sbuf1 = s1;

  for (int t = 0; t < T_STEPS; ++t) {
    // ---- device-wide barrier: step t-1 stores visible before step t reads ----
    __syncthreads();
    if (tid == 0) {
      unsigned g = __hip_atomic_load(&bar[1], __ATOMIC_ACQUIRE, __HIP_MEMORY_SCOPE_AGENT);
      unsigned a = __hip_atomic_fetch_add(&bar[0], 1u, __ATOMIC_ACQ_REL, __HIP_MEMORY_SCOPE_AGENT);
      if (a == (unsigned)nblocks - 1u) {
        __hip_atomic_store(&bar[0], 0u, __ATOMIC_RELAXED, __HIP_MEMORY_SCOPE_AGENT);
        __hip_atomic_store(&bar[1], g + 1u, __ATOMIC_RELEASE, __HIP_MEMORY_SCOPE_AGENT);
      } else {
        while (__hip_atomic_load(&bar[1], __ATOMIC_ACQUIRE, __HIP_MEMORY_SCOPE_AGENT) == g)
          __builtin_amdgcn_s_sleep(1);
      }
    }
    __syncthreads();

    // ---- async copy prev state [16 x 1024] bf16 (32 KB) into LDS ----
    {
      const uint16_t* src = ((t & 1) ? sbuf1 : sbuf0) + (size_t)bt * 16 * N_HID;
      #pragma unroll
      for (int it = 0; it < 8; ++it) {          // 2048 x 16B chunks / 256 threads
        unsigned chunk  = (unsigned)(it * 256 + tid);
        unsigned ldsoff = lds_state_base + chunk * 16u;
        unsigned goff   = chunk * 16u;
        asm volatile("global_load_async_to_lds_b128 %0, %1, %2 scope:SCOPE_DEV"
                     :: "v"(ldsoff), "v"(goff), "s"(src) : "memory");
      }
    }

    // ---- x[t] tile [16 x 256] f32 -> bf16 LDS (fused input projection) ----
    {
      const float4* xs = (const float4*)(x + ((size_t)t * BATCH + bt * 16) * N_INP);
      #pragma unroll
      for (int it = 0; it < 4; ++it) {
        int q = it * 256 + tid;
        float4 v = xs[q];
        int e = q * 4;
        lds_x[e + 0] = f2bf(v.x); lds_x[e + 1] = f2bf(v.y);
        lds_x[e + 2] = f2bf(v.z); lds_x[e + 3] = f2bf(v.w);
      }
    }

    asm volatile("s_wait_asynccnt 0" ::: "memory");
    __syncthreads();

    // ---- D = bias + x_t @ W_ih^T + state @ W_hh^T (f32 acc, bf16 WMMA) ----
    v8f acc;
    #pragma unroll
    for (int r = 0; r < 8; ++r) acc[r] = bias;

    {  // input projection: K = 256
      const uint16_t* bcol = wih_bf + (size_t)hcol * N_INP;
      #pragma unroll
      for (int kb = 0; kb < N_INP / 32; ++kb) {
        v16bf a = load_A(&lds_x[mrow * N_INP + kb * 32 + kgrp * 16]);
        v16bf b = load_B(&bcol[kb * 32 + kgrp * 16]);
        acc = __builtin_amdgcn_wmma_f32_16x16x32_bf16(false, a, false, b,
                                                      (short)0, acc, false, false);
      }
    }
    {  // recurrence: K = 1024, W_hh bf16 tile streamed from L2 (2 MB resident)
      const uint16_t* bcol = whh_bf + (size_t)hcol * N_HID;
      #pragma unroll 8
      for (int kb = 0; kb < N_HID / 32; ++kb) {
        v16bf a = load_A(&lds_state[mrow * N_HID + kb * 32 + kgrp * 16]);
        v16bf b = load_B(&bcol[kb * 32 + kgrp * 16]);
        acc = __builtin_amdgcn_wmma_f32_16x16x32_bf16(false, a, false, b,
                                                      (short)0, acc, false, false);
      }
    }

    // ---- tanh, publish new state (double-buffered), final state in f32 ----
    uint16_t* dst = ((t & 1) ? sbuf0 : sbuf1);  // write S[(t+1)&1]
    #pragma unroll
    for (int r = 0; r < 8; ++r) {               // C/D: M = r + 8*kgrp, N = mrow
      float h = tanhf(acc[r]);
      int brow = bt * 16 + r + 8 * kgrp;
      dst[(size_t)brow * N_HID + hcol] = f2bf(h);
      if (t == T_STEPS - 1) fin[(size_t)brow * N_HID + hcol] = h;
    }
    __threadfence();
  }
}

// ---------------- readout: out = fin @ W_ro^T + b_ro  (WMMA, 8x8 tiles) -----
__global__ void readout_kernel(const float* __restrict__ fin,
                               const float* __restrict__ W_ro,
                               const float* __restrict__ b_ro,
                               float* __restrict__ out) {
  const int lane = threadIdx.x & 31;
  const int mt = blockIdx.x >> 3, nt = blockIdx.x & 7;
  const int mrow = lane & 15, kgrp = lane >> 4;
  const int ocol = nt * 16 + mrow;
  const float bias = b_ro[ocol];
  v8f acc;
  #pragma unroll
  for (int r = 0; r < 8; ++r) acc[r] = bias;

  const float* arow = fin  + (size_t)(mt * 16 + mrow) * N_HID;
  const float* brow = W_ro + (size_t)ocol * N_HID;
  for (int kb = 0; kb < N_HID / 32; ++kb) {
    PackCast a, b;
    int ka = kb * 32 + kgrp * 16;   // A: contiguous K pattern
    int kbase = kb * 32 + kgrp * 16; // B: contiguous K pattern
    #pragma unroll
    for (int j = 0; j < 8; ++j) {
      a.s[j]     = f2bf(arow[ka + j]);
      a.s[8 + j] = f2bf(arow[ka + 8 + j]);
      b.s[j]     = f2bf(brow[kbase + j]);
      b.s[8 + j] = f2bf(brow[kbase + 8 + j]);
    }
    acc = __builtin_amdgcn_wmma_f32_16x16x32_bf16(false, a.v, false, b.v,
                                                  (short)0, acc, false, false);
  }
  #pragma unroll
  for (int r = 0; r < 8; ++r)
    out[(size_t)(mt * 16 + r + 8 * kgrp) * N_OUT + ocol] = acc[r];
}

// ---------------- launch -----------------------------------------------------
extern "C" void kernel_launch(void* const* d_in, const int* in_sizes, int n_in,
                              void* d_out, int out_size, void* d_ws, size_t ws_size,
                              hipStream_t stream) {
  const float* x    = (const float*)d_in[0];
  const float* W_ih = (const float*)d_in[1];
  const float* W_hh = (const float*)d_in[2];
  const float* b_ih = (const float*)d_in[3];
  const float* b_hh = (const float*)d_in[4];
  const float* W_ro = (const float*)d_in[5];
  const float* b_ro = (const float*)d_in[6];
  float* out = (float*)d_out;

  char* ws = (char*)d_ws;                          // ~3.7 MB total
  uint16_t* whh_bf = (uint16_t*)(ws + 0);          // 2 MB
  uint16_t* wih_bf = (uint16_t*)(ws + 2097152);    // 512 KB
  uint16_t* s0     = (uint16_t*)(ws + 2621440);    // 256 KB state staging A
  uint16_t* s1     = (uint16_t*)(ws + 2883584);    // 256 KB state staging B
  float*    fin    = (float*)   (ws + 3145728);    // 512 KB final state f32
  unsigned* bar    = (unsigned*)(ws + 3670016);    // barrier counter + generation

  prep_kernel<<<512, 256, 0, stream>>>(W_ih, W_hh, whh_bf, wih_bf, s0, bar);
  rnn_persist<<<64, 256, 0, stream>>>(x, whh_bf, wih_bf, b_ih, b_hh, s0, s1, fin, bar);
  readout_kernel<<<64, 32, 0, stream>>>(fin, W_ro, b_ro, out);
}
